// TverskyLayer_29497835389698
// MI455X (gfx1250) — matrix operand order repaired and checked by reading, not compiled, not run on hardware
//
#include <hip/hip_runtime.h>

typedef __bf16 bf16;
typedef bf16  bf16x8  __attribute__((ext_vector_type(8)));
typedef bf16  bf16x16 __attribute__((ext_vector_type(16)));
typedef float f32x8   __attribute__((ext_vector_type(8)));

// Problem dims (fixed by the reference).
constexpr int Bn = 4096;   // batch rows
constexpr int Dd = 1024;   // feature dim
constexpr int Ff = 2048;   // num features
constexpr int Pp = 1024;   // num prototypes

// LDS tile row strides (elements). 72*2B = 144B = 9*16B: keeps b128 alignment
// and spreads 16 row-strided lanes over all 64 LDS banks.
constexpr int LDA_S = 72;
constexpr int LDB_S = 72;

static __device__ __forceinline__ bf16 f32_to_bf16(float f) {
  unsigned u = __builtin_bit_cast(unsigned, f);
  u += 0x7FFFu + ((u >> 16) & 1u);          // round-to-nearest-even
  unsigned short h = (unsigned short)(u >> 16);
  return __builtin_bit_cast(bf16, h);
}
static __device__ __forceinline__ float bf16_to_f32(bf16 b) {
  unsigned short h = __builtin_bit_cast(unsigned short, b);
  unsigned u = ((unsigned)h) << 16;
  return __builtin_bit_cast(float, u);
}

// ---- async global -> LDS copy (16B per lane), tracked by ASYNCcnt --------
static __device__ __forceinline__ void async_copy16(bf16* lds_p, const bf16* gp) {
  // Generic pointer to LDS: low 32 bits are the LDS byte address (ISA 10.2).
  unsigned la = (unsigned)(unsigned long long)lds_p;
  asm volatile("global_load_async_to_lds_b128 %0, %1, off"
               :: "v"(la), "v"(gp) : "memory");
}

// ---- WMMA fragment loads (wave32, 16x16x32 bf16) from LDS tiles ----------
// A (16x32, MxK): lanes 0-15 -> M=lane, K = h*8 + [0..8) and 16 + h*8 + [0..8)
static __device__ __forceinline__ bf16x16 load_a_frag(const bf16* __restrict__ A,
                                                      int row, int k, int ld) {
  int lane = (int)(threadIdx.x & 31u);
  int r = lane & 15, h = lane >> 4;
  const bf16* p = A + (size_t)(row + r) * (size_t)ld + (size_t)(k + h * 8);
  bf16x8 lo = *(const bf16x8*)p;
  bf16x8 hi = *(const bf16x8*)(p + 16);
  bf16x16 f;
#pragma unroll
  for (int i = 0; i < 8; ++i) { f[i] = lo[i]; f[i + 8] = hi[i]; }
  return f;
}
// B (32x16, KxN) sourced from row-major [N x K]: lane -> N=lane&15,
// K = h*8 + [0..8) and 16 + h*8 + [0..8).
static __device__ __forceinline__ bf16x16 load_b_frag(const bf16* __restrict__ Bm,
                                                      int col, int k, int ld) {
  int lane = (int)(threadIdx.x & 31u);
  int r = lane & 15, h = lane >> 4;
  const bf16* p = Bm + (size_t)(col + r) * (size_t)ld + (size_t)(k + h * 8);
  bf16x8 lo = *(const bf16x8*)p;
  bf16x8 hi = *(const bf16x8*)(p + 16);
  bf16x16 f;
#pragma unroll
  for (int i = 0; i < 8; ++i) { f[i] = lo[i]; f[i + 8] = hi[i]; }
  return f;
}

static __device__ __forceinline__ f32x8 wmma_bf16(bf16x16 a, bf16x16 b, f32x8 c) {
  return __builtin_amdgcn_wmma_f32_16x16x32_bf16(
      /*neg_a=*/false, a, /*neg_b=*/false, b,
      /*c_mod=*/(short)0, c, /*reuse_a=*/false, /*reuse_b=*/false);
}

// Issue one KT=64 stage: A tile 64x64 (2 chunks/thread), B tile 128x64 (4 chunks/thread).
static __device__ __forceinline__ void issue_stage(
    const bf16* __restrict__ A, const bf16* __restrict__ Bm,
    int lda, int ldb, int m0, int n0, int k0, bf16* sA, bf16* sB) {
  int t = (int)threadIdx.x;
#pragma unroll
  for (int i = 0; i < 2; ++i) {
    int id = t + i * 256;
    int row = id >> 3, c = (id & 7) * 8;
    async_copy16(sA + row * LDA_S + c,
                 A + (size_t)(m0 + row) * (size_t)lda + (size_t)(k0 + c));
  }
#pragma unroll
  for (int i = 0; i < 4; ++i) {
    int id = t + i * 256;
    int row = id >> 3, c = (id & 7) * 8;
    async_copy16(sB + row * LDB_S + c,
                 Bm + (size_t)(n0 + row) * (size_t)ldb + (size_t)(k0 + c));
  }
}

// Shared double-buffered WMMA core. Each wave (of 8, 2Mx4N) owns a 32x32 tile.
#define GEMM_CORE(A_, B_, K_, lda_, ldb_, m0_, n0_, acc_)                       \
  __shared__ bf16 sA[2][64 * LDA_S];                                            \
  __shared__ bf16 sB[2][128 * LDB_S];                                           \
  int wave = (int)(threadIdx.x >> 5);                                           \
  int wm = wave & 1, wn = wave >> 1;                                            \
  int mw = wm * 32, nw = wn * 32;                                               \
  int nk = (K_) / 64;                                                           \
  issue_stage((A_), (B_), (lda_), (ldb_), (m0_), (n0_), 0, sA[0], sB[0]);       \
  for (int kt = 0; kt < nk; ++kt) {                                             \
    if (kt + 1 < nk) {                                                          \
      issue_stage((A_), (B_), (lda_), (ldb_), (m0_), (n0_), (kt + 1) * 64,      \
                  sA[(kt + 1) & 1], sB[(kt + 1) & 1]);                          \
      asm volatile("s_wait_asynccnt 6" ::: "memory");                           \
    } else {                                                                    \
      asm volatile("s_wait_asynccnt 0" ::: "memory");                           \
    }                                                                           \
    __syncthreads();                                                            \
    const bf16* sAb = sA[kt & 1];                                               \
    const bf16* sBb = sB[kt & 1];                                               \
    _Pragma("unroll")                                                           \
    for (int kk = 0; kk < 64; kk += 32) {                                       \
      bf16x16 a0 = load_a_frag(sAb, mw,      kk, LDA_S);                        \
      bf16x16 a1 = load_a_frag(sAb, mw + 16, kk, LDA_S);                        \
      bf16x16 b0 = load_b_frag(sBb, nw,      kk, LDB_S);                        \
      bf16x16 b1 = load_b_frag(sBb, nw + 16, kk, LDB_S);                        \
      acc_[0][0] = wmma_bf16(a0, b0, acc_[0][0]);                               \
      acc_[0][1] = wmma_bf16(a0, b1, acc_[0][1]);                               \
      acc_[1][0] = wmma_bf16(a1, b0, acc_[1][0]);                               \
      acc_[1][1] = wmma_bf16(a1, b1, acc_[1][1]);                               \
    }                                                                           \
    __syncthreads();                                                            \
  }

// ---- fp32 -> bf16 conversion ---------------------------------------------
__global__ __launch_bounds__(256) void convert_bf16_kernel(
    const float* __restrict__ in, bf16* __restrict__ out, int n) {
  int i = blockIdx.x * 256 + threadIdx.x;
  if (i < n) out[i] = f32_to_bf16(in[i]);
}

// ---- Phase 1: T = A @ Bm^T ; out0 = bf16(T*sigmoid(T)) ; out1 = bf16(sigmoid(T))
__global__ __launch_bounds__(256) void gemm_sig_kernel(
    const bf16* __restrict__ A, const bf16* __restrict__ Bm,
    bf16* __restrict__ out0, bf16* __restrict__ out1,
    int K, int lda, int ldb, int ldo) {
  int m0 = blockIdx.y * 64;
  int n0 = blockIdx.x * 128;
  f32x8 acc[2][2] = {};
  GEMM_CORE(A, Bm, K, lda, ldb, m0, n0, acc)
  int lane = (int)(threadIdx.x & 31u);
  int r = lane & 15, h = lane >> 4;
#pragma unroll
  for (int mi = 0; mi < 2; ++mi)
#pragma unroll
    for (int ni = 0; ni < 2; ++ni)
#pragma unroll
      for (int i = 0; i < 8; ++i) {
        float t = acc[mi][ni][i];
        float s = 1.0f / (1.0f + __expf(-t));
        float a = t * s;
        size_t gr = (size_t)(m0 + mw + mi * 16 + h * 8 + i);
        size_t gc = (size_t)(n0 + nw + ni * 16 + r);
        out0[gr * (size_t)ldo + gc] = f32_to_bf16(a);
        out1[gr * (size_t)ldo + gc] = f32_to_bf16(s);
      }
}

// ---- Final fused GEMM: out = Acat@Bcat^T - alpha*rsA[b] - beta*rsB[p] ----
__global__ __launch_bounds__(256) void gemm_final_kernel(
    const bf16* __restrict__ A, const bf16* __restrict__ Bm,
    const float* __restrict__ rsA, const float* __restrict__ rsB,
    const float* __restrict__ alphaP, const float* __restrict__ betaP,
    float* __restrict__ out, int K, int lda, int ldb, int N) {
  int m0 = blockIdx.y * 64;
  int n0 = blockIdx.x * 128;
  f32x8 acc[2][2] = {};
  GEMM_CORE(A, Bm, K, lda, ldb, m0, n0, acc)
  float alpha = alphaP[0], beta = betaP[0];
  int lane = (int)(threadIdx.x & 31u);
  int r = lane & 15, h = lane >> 4;
#pragma unroll
  for (int mi = 0; mi < 2; ++mi)
#pragma unroll
    for (int ni = 0; ni < 2; ++ni)
#pragma unroll
      for (int i = 0; i < 8; ++i) {
        int gr = m0 + mw + mi * 16 + h * 8 + i;
        int gc = n0 + nw + ni * 16 + r;
        float t = acc[mi][ni][i];
        out[(size_t)gr * (size_t)N + (size_t)gc] =
            t - alpha * rsA[gr] - beta * rsB[gc];
      }
}

// ---- Row sums of the first Ff columns (bf16 -> f32) ----------------------
__global__ __launch_bounds__(256) void rowsum_kernel(
    const bf16* __restrict__ src, float* __restrict__ dst, int width, int ld) {
  __shared__ float red[256];
  int row = blockIdx.x;
  float s = 0.0f;
  for (int f = (int)threadIdx.x; f < width; f += 256)
    s += bf16_to_f32(src[(size_t)row * (size_t)ld + (size_t)f]);
  red[threadIdx.x] = s;
  __syncthreads();
  for (int off = 128; off > 0; off >>= 1) {
    if ((int)threadIdx.x < off) red[threadIdx.x] += red[threadIdx.x + off];
    __syncthreads();
  }
  if (threadIdx.x == 0) dst[row] = red[0];
}

// ---- In-place rescale: [bp | sp] -> [theta*bp + alpha*sp | beta*bp] ------
__global__ __launch_bounds__(256) void scale_pack_kernel(
    bf16* __restrict__ Bcat, const float* __restrict__ thetaP,
    const float* __restrict__ alphaP, const float* __restrict__ betaP) {
  int idx = blockIdx.x * 256 + threadIdx.x;          // over Pp*Ff
  int p = idx / Ff, f = idx - p * Ff;
  size_t base = (size_t)p * (size_t)(2 * Ff);
  float bp = bf16_to_f32(Bcat[base + f]);
  float sp = bf16_to_f32(Bcat[base + Ff + f]);
  float th = thetaP[0], al = alphaP[0], be = betaP[0];
  Bcat[base + f]      = f32_to_bf16(th * bp + al * sp);
  Bcat[base + Ff + f] = f32_to_bf16(be * bp);
}

extern "C" void kernel_launch(void* const* d_in, const int* in_sizes, int n_in,
                              void* d_out, int out_size, void* d_ws, size_t ws_size,
                              hipStream_t stream) {
  (void)in_sizes; (void)n_in; (void)out_size; (void)ws_size;
  const float* x          = (const float*)d_in[0];
  const float* features   = (const float*)d_in[1];
  const float* prototypes = (const float*)d_in[2];
  const float* alphaP     = (const float*)d_in[3];
  const float* betaP      = (const float*)d_in[4];
  const float* thetaP     = (const float*)d_in[5];
  float* out = (float*)d_out;

  char* ws = (char*)d_ws;
  size_t off = 0;
  auto alloc = [&](size_t bytes) -> void* {
    void* p = (void*)(ws + off);
    off += (bytes + 255u) & ~(size_t)255u;
    return p;
  };
  bf16*  xb   = (bf16*)alloc((size_t)Bn * Dd * sizeof(bf16));       // 8 MiB
  bf16*  fb   = (bf16*)alloc((size_t)Ff * Dd * sizeof(bf16));       // 4 MiB
  bf16*  pb   = (bf16*)alloc((size_t)Pp * Dd * sizeof(bf16));       // 2 MiB
  bf16*  Acat = (bf16*)alloc((size_t)Bn * 2 * Ff * sizeof(bf16));   // 32 MiB  [ax | sx]
  bf16*  Bcat = (bf16*)alloc((size_t)Pp * 2 * Ff * sizeof(bf16));   // 8 MiB   [bp | sp] -> [u | v]
  float* rsA  = (float*)alloc((size_t)Bn * sizeof(float));
  float* rsB  = (float*)alloc((size_t)Pp * sizeof(float));

  // 1) fp32 -> bf16 operand conversion
  convert_bf16_kernel<<<dim3((Bn * Dd) / 256), 256, 0, stream>>>(x, xb, Bn * Dd);
  convert_bf16_kernel<<<dim3((Ff * Dd) / 256), 256, 0, stream>>>(features, fb, Ff * Dd);
  convert_bf16_kernel<<<dim3((Pp * Dd) / 256), 256, 0, stream>>>(prototypes, pb, Pp * Dd);

  // 2) Projection GEMMs + sigmoid epilogue (async-LDS double-buffered WMMA)
  gemm_sig_kernel<<<dim3(Ff / 128, Bn / 64), 256, 0, stream>>>(
      xb, fb, Acat, Acat + Ff, Dd, Dd, Dd, 2 * Ff);
  gemm_sig_kernel<<<dim3(Ff / 128, Pp / 64), 256, 0, stream>>>(
      pb, fb, Bcat, Bcat + Ff, Dd, Dd, Dd, 2 * Ff);

  // 3) Row sums (before Bcat is rescaled)
  rowsum_kernel<<<dim3(Bn), 256, 0, stream>>>(Acat, rsA, Ff, 2 * Ff);
  rowsum_kernel<<<dim3(Pp), 256, 0, stream>>>(Bcat, rsB, Ff, 2 * Ff);

  // 4) [bp | sp] -> [theta*bp + alpha*sp | beta*bp]
  scale_pack_kernel<<<dim3((Pp * Ff) / 256), 256, 0, stream>>>(Bcat, thetaP, alphaP, betaP);

  // 5) Single fused output GEMM (K = 2F = 4096) + row-sum corrections
  gemm_final_kernel<<<dim3(Pp / 128, Bn / 64), 256, 0, stream>>>(
      Acat, Bcat, rsA, rsB, alphaP, betaP, out, 2 * Ff, 2 * Ff, 2 * Ff, Pp);
}
